// FlatMixedEffectsDecoder_25211458028101
// MI455X (gfx1250) — matrix operand order repaired and checked by reading, not compiled, not run on hardware
//
#include <hip/hip_runtime.h>

// Problem constants (match the reference).
constexpr int V = 32000, M = 512, P = 16, R = 8, B = 512;

// Tiling for the logits kernel.
constexpr int ROWS    = 16;   // batch rows per block (one WMMA M tile)
constexpr int COLS    = 256;  // vocab columns per block (16 N tiles)
constexpr int THREADS = 256;  // 8 wave32s; 2 column tiles per wave
constexpr int KTOT    = P * R; // 128 = 4 x K32 WMMA steps

typedef __attribute__((ext_vector_type(16))) _Float16 v16h;
typedef __attribute__((ext_vector_type(8)))  float    v8f;

// Monotonic float<->int mapping so integer atomicMax == float max (exact, order-independent).
__device__ __forceinline__ int f2ord(float f) {
    int i = __float_as_int(f);
    return (i >= 0) ? i : (i ^ 0x7fffffff);
}
__device__ __forceinline__ float ord2f(int i) {
    return __int_as_float((i >= 0) ? i : (i ^ 0x7fffffff));
}

__global__ void init_rowmax_kernel(int* __restrict__ rowmax_i) {
    if (threadIdx.x < (unsigned)B) rowmax_i[threadIdx.x] = (int)0x80000000; // INT_MIN
}

// Pass 1: logits[b,v] = bg + author + (z_b . eta_persona[:, r_b, v]) via WMMA over K=P*R,
// staged to d_out; per-row running max -> rowmax_i (mapped-int atomicMax).
__global__ __launch_bounds__(THREADS)
void logits_kernel(const int*   __restrict__ m_idx,
                   const float* __restrict__ z_persona,
                   const int*   __restrict__ r_idx,
                   const float* __restrict__ eta_bg,
                   const float* __restrict__ eta_author,
                   const float* __restrict__ eta_persona,
                   float*       __restrict__ logits,     // staging = d_out [B][V]
                   int*         __restrict__ rowmax_i)   // [B], mapped ints
{
    __shared__ int lmax[ROWS];

    const int tid  = threadIdx.x;
    const int wave = tid >> 5;
    const int lane = tid & 31;
    const int hi   = lane >> 4;   // half-wave select
    const int ln   = lane & 15;   // 0..15
    const int b0   = blockIdx.y * ROWS;
    const int c0   = blockIdx.x * COLS;

    if (tid < ROWS) lmax[tid] = (int)0x80000000;
    __syncthreads();

    // ---- Build A' fragments in registers (row = ln for every lane).
    // A'[row, k] with k = r*16 + p; fragment step s covers K = 32s..32s+31.
    // Lane layout (16-bit A 16x32): half h -> K = 32s + (h&7) + 16*(h>=8) + 8*hi,
    // hence p = (h&7) + 8*hi and r = 2s + (h>=8).
    const int   row_b = b0 + ln;
    const int   r_row = r_idx[row_b];
    const float* zp   = z_persona + row_b * P + 8 * hi;
    float zv[8];
#pragma unroll
    for (int j = 0; j < 8; ++j) zv[j] = zp[j];

    v16h a[4];
#pragma unroll
    for (int s = 0; s < 4; ++s) {
#pragma unroll
        for (int h = 0; h < 16; ++h) {
            const int r_of_h = 2 * s + (h >> 3);
            a[s][h] = (r_of_h == r_row) ? (_Float16)zv[h & 7] : (_Float16)0.0f;
        }
    }

    // Per-output-row gather indices (shared by both column tiles).
    const int rbase = hi * 8;
    int rr[8], mm[8];
#pragma unroll
    for (int j = 0; j < 8; ++j) {
        const int rw = b0 + rbase + j;
        rr[j] = r_idx[rw];
        mm[j] = m_idx[rw];
    }

    // ---- Two 16-column tiles per wave.
#pragma unroll
    for (int t = 0; t < 2; ++t) {
        const int tile = wave * 2 + t;
        const int col  = c0 + tile * 16 + ln;

        v8f c = 0.0f;
#pragma unroll
        for (int s = 0; s < 4; ++s) {
            // B'[k, col] = eta_persona[(p*R + r)*V + col]; half h -> p = (h&7) + 8*hi, r = 2s + (h>=8).
            const int    r_b = 2 * s;
            const float* bp  = eta_persona + (size_t)r_b * V + col;
            v16h bfrag;
#pragma unroll
            for (int h = 0; h < 16; ++h)
                bfrag[h] = (_Float16)bp[((size_t)((h & 7) + 8 * hi) * R + (h >> 3)) * V];

            c = __builtin_amdgcn_wmma_f32_16x16x32_f16(
                    false, a[s], false, bfrag, (short)0, c, false, false);
        }

        // C layout: VGPR j holds row (rbase + j), column = col for this lane.
#pragma unroll
        for (int j = 0; j < 8; ++j) {
            const int rw = b0 + rbase + j;
            float logit = c[j]
                        + eta_bg[(size_t)rr[j] * V + col]
                        + eta_author[((size_t)mm[j] * R + rr[j]) * V + col];
            logits[(size_t)rw * V + col] = logit;

            // Row max across the 16 lanes of this half-wave (masks < 16 stay in-half).
            float mx = logit;
#pragma unroll
            for (int off = 1; off < 16; off <<= 1)
                mx = fmaxf(mx, __shfl_xor(mx, off, 32));
            if (ln == 0) atomicMax(&lmax[rbase + j], f2ord(mx));
        }
    }

    __syncthreads();
    if (tid < ROWS) atomicMax(&rowmax_i[b0 + tid], lmax[tid]);
}

// Pass 2: one block per row; deterministic tree reduction of sum(exp(x - max)).
__global__ __launch_bounds__(256)
void rowstat_kernel(const float* __restrict__ logits,
                    const int*   __restrict__ rowmax_i,
                    float*       __restrict__ rowstat)
{
    __shared__ float red[256];
    const int    b   = blockIdx.x;
    const float  mx  = ord2f(rowmax_i[b]);
    const float* row = logits + (size_t)b * V;

    float s = 0.0f;
    for (int v = threadIdx.x; v < V; v += 256)
        s += expf(row[v] - mx);

    red[threadIdx.x] = s;
    __syncthreads();
    for (int off = 128; off > 0; off >>= 1) {
        if ((int)threadIdx.x < off) red[threadIdx.x] += red[threadIdx.x + off];
        __syncthreads();
    }
    if (threadIdx.x == 0) rowstat[b] = mx + logf(red[0]);
}

// Pass 3: out[b,v] = logit - (max + log(sumexp)); float4 RMW.
__global__ __launch_bounds__(256)
void finalize_kernel(float* __restrict__ out, const float* __restrict__ rowstat)
{
    const size_t i       = (size_t)blockIdx.x * blockDim.x + threadIdx.x;
    const int    per_row = V / 4; // 8000, V % 4 == 0
    if (i >= (size_t)B * per_row) return;
    float4* o4 = (float4*)out;
    const int   b  = (int)(i / per_row);
    const float st = rowstat[b];
    float4 vv = o4[i];
    vv.x -= st; vv.y -= st; vv.z -= st; vv.w -= st;
    o4[i] = vv;
}

extern "C" void kernel_launch(void* const* d_in, const int* in_sizes, int n_in,
                              void* d_out, int out_size, void* d_ws, size_t ws_size,
                              hipStream_t stream) {
    (void)in_sizes; (void)n_in; (void)out_size; (void)ws_size;

    const int*   m_idx = (const int*)  d_in[0];
    const float* z     = (const float*)d_in[1];
    const int*   r_idx = (const int*)  d_in[2];
    const float* ebg   = (const float*)d_in[3];
    const float* eau   = (const float*)d_in[4];
    const float* epe   = (const float*)d_in[5];
    float* out = (float*)d_out;

    int*   rowmax_i = (int*)d_ws;                                  // B ints
    float* rowstat  = (float*)((char*)d_ws + B * sizeof(int));     // B floats

    init_rowmax_kernel<<<1, 512, 0, stream>>>(rowmax_i);

    dim3 g1(V / COLS, B / ROWS); // 125 x 32 = 4000 blocks
    logits_kernel<<<g1, THREADS, 0, stream>>>(m_idx, z, r_idx, ebg, eau, epe, out, rowmax_i);

    rowstat_kernel<<<B, 256, 0, stream>>>(out, rowmax_i, rowstat);

    finalize_kernel<<<(B * (V / 4)) / 256, 256, 0, stream>>>(out, rowstat);
}
